// SynthesisBlockAttention_34737695490206
// MI455X (gfx1250) — matrix operand order repaired and checked by reading, not compiled, not run on hardware
//
#include <hip/hip_runtime.h>
#include <hip/hip_bf16.h>
#include <cstddef>

// ---------------------------------------------------------------------------
// Shapes (fixed by the reference):
//   B=8, C=64, IMG=64, P=8, NPA=8, N=64, D=4096, WDIM=512, T=64, TDIM=512
// ---------------------------------------------------------------------------

typedef __attribute__((ext_vector_type(16))) __bf16 v16bf;
typedef __attribute__((ext_vector_type(2)))  __bf16 v2bf;
typedef __attribute__((ext_vector_type(8)))  float  v8f;

// Pack two fp32 -> two bf16 in one dword: low16 = bf16(a), high16 = bf16(b).
// Round-half-up (+0x8000) then v_perm_b32 byte-select of the high halves.
__device__ __forceinline__ unsigned pkbf(float a, float b) {
    unsigned ua = __builtin_bit_cast(unsigned, a) + 0x8000u;
    unsigned ub = __builtin_bit_cast(unsigned, b) + 0x8000u;
    // concat = {S0=ub (bytes 4-7), S1=ua (bytes 0-3)}; select 2,3,6,7
    return __builtin_amdgcn_perm(ub, ua, 0x07060302u);
}

// ---------------------------------------------------------------------------
// Kernel 1: patchify x (NCHW) into hs[b, n, d] and append style row n==64:
//   style = w @ sa_style_w + sa_style_b
// hs layout: (B, 65, 4096) contiguous; thread t == flat index.
// ---------------------------------------------------------------------------
__global__ __launch_bounds__(256) void build_hs(const float* __restrict__ x,
                                                const float* __restrict__ w,
                                                const float* __restrict__ sw,
                                                const float* __restrict__ sb,
                                                float* __restrict__ hs) {
    int t = blockIdx.x * 256 + threadIdx.x;
    int d = t & 4095;
    int n = (t >> 12) % 65;
    int b = t / (65 * 4096);
    float val;
    if (n < 64) {
        int ib = n >> 3, jb = n & 7;
        int c = d & 63, pp = d >> 6, pc = pp & 7, pr = pp >> 3;
        int hh = jb * 8 + pr, ww = ib * 8 + pc;
        val = x[(((size_t)(b * 64 + c)) * 64 + hh) * 64 + ww];
    } else {
        float a = sb[d];
        const float* wr = w + b * 512;
        for (int kk = 0; kk < 512; ++kk) a += wr[kk] * sw[(size_t)kk * 4096 + d];
        val = a;
    }
    hs[t] = val;
}

// ---------------------------------------------------------------------------
// Generic GEMM: Cc = act(A(MxK) @ W(KxN) + bias), fp32 in/out, bf16 WMMA core.
// A rows are remapped:  r -> (r / rpbU) * rpbA + (r % rpbU).
// Output rows contiguous.  slope: 1.0 = identity, 0.01 = leaky ReLU.
//
// Block tile 128(M) x 64(N), K-step 32; 8 waves (4x2), each wave 2x2 16x16.
// LDS double-buffered; tiles stored in WMMA fragment-element order so each
// fragment is two contiguous ds_load_b128 (no repacking):
//   A LDS row m, slot p:  k = kgroup[p>>3] + (p&7),  kgroup = {0,8,16,24}
//     -> lane fragment (16 elems) = sA[m][16*half .. 16*half+15]
//   B LDS transposed [n][k] -> lane fragment = sB[n][16*half .. 16*half+15]
// Stride 48 elems => all fragment starts are 32B aligned.
// ---------------------------------------------------------------------------
#define BM 128
#define BN 64
#define BK 32
#define LSTR 48   // LDS stride (elements) for both A and B tiles

__global__ __launch_bounds__(256) void gemm_bf16(const float* __restrict__ A,
                                                 const float* __restrict__ W,
                                                 const float* __restrict__ bias,
                                                 float* __restrict__ Cc,
                                                 int M, int N, int K,
                                                 int rpbU, int rpbA, float slope) {
    __shared__ __bf16 sA[2][BM * LSTR];   // 2 x 12288 B
    __shared__ __bf16 sB[2][BN * LSTR];   // 2 x  6144 B

    const int tid  = threadIdx.x;
    const int lane = tid & 31;
    const int wave = tid >> 5;
    const int half = lane >> 4;
    const int lm   = lane & 15;
    const int wm0  = (wave & 3) << 5;     // 0,32,64,96
    const int wn0  = (wave >> 2) << 5;    // 0,32
    const int m0   = blockIdx.y * BM;
    const int n0   = blockIdx.x * BN;

    // ---- staging descriptors -------------------------------------------
    // A: 128x32 floats = 1024 quads; 4 quads/thread.
    const int kgroup[4] = {0, 8, 16, 24};
    int aOff[4], aLds[4];
    bool aVal[4];
#pragma unroll
    for (int i = 0; i < 4; ++i) {
        int quad = tid + i * 256;
        int m = quad >> 3, q = quad & 7;
        int gRow = m0 + m;
        aVal[i] = gRow < M;
        int rc = aVal[i] ? gRow : (M - 1);             // clamped, always legal
        int mr = (rc / rpbU) * rpbA + (rc % rpbU);
        int col = kgroup[q >> 1] + (q & 1) * 4;        // contiguous 4 floats
        aOff[i] = mr * K + col;
        aLds[i] = m * LSTR + q * 4;
    }
    // B: 32x64 floats = 512 quads; 2 quads/thread (transposed store).
    int bOff[2], bK[2], bN0[2];
#pragma unroll
    for (int i = 0; i < 2; ++i) {
        int quad = tid + i * 256;
        int k = quad >> 4, q = quad & 15;
        bK[i]  = k;
        bN0[i] = q * 4;
        bOff[i] = k * N + n0 + q * 4;
    }

    // ---- accumulators ---------------------------------------------------
    v8f acc[2][2];
#pragma unroll
    for (int mt = 0; mt < 2; ++mt)
#pragma unroll
        for (int nt = 0; nt < 2; ++nt)
#pragma unroll
            for (int r = 0; r < 8; ++r) acc[mt][nt][r] = 0.0f;

    float4 rA[4], rB[2];
    const int ntiles = K >> 5;

    // prologue: fetch tile 0
#pragma unroll
    for (int i = 0; i < 4; ++i) {
        float4 v = *(const float4*)(A + aOff[i]);
        v.x = aVal[i] ? v.x : 0.0f;  v.y = aVal[i] ? v.y : 0.0f;
        v.z = aVal[i] ? v.z : 0.0f;  v.w = aVal[i] ? v.w : 0.0f;
        rA[i] = v;
    }
#pragma unroll
    for (int i = 0; i < 2; ++i) rB[i] = *(const float4*)(W + bOff[i]);

    int buf = 0;
    for (int kt = 0; kt < ntiles; ++kt) {
        // ---- store staged regs -> LDS[buf] (v_perm packed bf16) ----
#pragma unroll
        for (int i = 0; i < 4; ++i) {
            uint2 u;
            u.x = pkbf(rA[i].x, rA[i].y);
            u.y = pkbf(rA[i].z, rA[i].w);
            *(uint2*)&sA[buf][aLds[i]] = u;            // ds_store_b64
        }
#pragma unroll
        for (int i = 0; i < 2; ++i) {
            v2bf p0 = __builtin_bit_cast(v2bf, pkbf(rB[i].x, rB[i].y));
            v2bf p1 = __builtin_bit_cast(v2bf, pkbf(rB[i].z, rB[i].w));
            sB[buf][(bN0[i] + 0) * LSTR + bK[i]] = p0[0];
            sB[buf][(bN0[i] + 1) * LSTR + bK[i]] = p0[1];
            sB[buf][(bN0[i] + 2) * LSTR + bK[i]] = p1[0];
            sB[buf][(bN0[i] + 3) * LSTR + bK[i]] = p1[1];
        }
        __syncthreads();

        // ---- prefetch next tile into regs (overlaps with WMMA) ----
        if (kt + 1 < ntiles) {
            const int k0 = (kt + 1) << 5;
#pragma unroll
            for (int i = 0; i < 4; ++i) {
                float4 v = *(const float4*)(A + aOff[i] + k0);
                v.x = aVal[i] ? v.x : 0.0f;  v.y = aVal[i] ? v.y : 0.0f;
                v.z = aVal[i] ? v.z : 0.0f;  v.w = aVal[i] ? v.w : 0.0f;
                rA[i] = v;
            }
#pragma unroll
            for (int i = 0; i < 2; ++i)
                rB[i] = *(const float4*)(W + (size_t)k0 * N + bOff[i]);
        }

        // ---- WMMA from LDS[buf]: fragments are contiguous 32B loads ----
        v16bf afr[2], bfr[2];
#pragma unroll
        for (int mt = 0; mt < 2; ++mt) {
            int m = wm0 + mt * 16 + lm;
            afr[mt] = *(const v16bf*)&sA[buf][m * LSTR + 16 * half];
        }
#pragma unroll
        for (int nt = 0; nt < 2; ++nt) {
            int n = wn0 + nt * 16 + lm;
            bfr[nt] = *(const v16bf*)&sB[buf][n * LSTR + 16 * half];
        }
#pragma unroll
        for (int mt = 0; mt < 2; ++mt)
#pragma unroll
            for (int nt = 0; nt < 2; ++nt)
                acc[mt][nt] = __builtin_amdgcn_wmma_f32_16x16x32_bf16(
                    false, afr[mt], false, bfr[nt], (short)0, acc[mt][nt],
                    false, false);
        buf ^= 1;
    }

    // C layout: VGPR r -> row r + 8*half, col = lane&15.
#pragma unroll
    for (int mt = 0; mt < 2; ++mt)
#pragma unroll
        for (int nt = 0; nt < 2; ++nt)
#pragma unroll
            for (int r = 0; r < 8; ++r) {
                int gm = m0 + wm0 + mt * 16 + r + 8 * half;
                int gn = n0 + wn0 + nt * 16 + lm;
                if (gm < M) {
                    float v = acc[mt][nt][r] + bias[gn];
                    Cc[(size_t)gm * N + gn] = v > 0.0f ? v : v * slope;
                }
            }
}

// ---------------------------------------------------------------------------
// kn[row] = sum_d K[row,d]^2      (one block per row, D=4096)
// ---------------------------------------------------------------------------
__global__ __launch_bounds__(256) void row_sqnorm(const float* __restrict__ X,
                                                  float* __restrict__ kn, int D_) {
    int row = blockIdx.x, tid = threadIdx.x;
    __shared__ float red[256];
    float a = 0.0f;
    for (int d = tid; d < D_; d += 256) {
        float v = X[(size_t)row * D_ + d];
        a += v * v;
    }
    red[tid] = a;
    __syncthreads();
    for (int s = 128; s > 0; s >>= 1) {
        if (tid < s) red[tid] += red[tid + s];
        __syncthreads();
    }
    if (tid == 0) kn[row] = red[0];
}

// ---------------------------------------------------------------------------
// Fused scores + softmax per (b, q):
//   wts[b,q,j] = softmax_j( 2 * q.k_j - ||k_j||^2 )   (||q||^2 cancels)
// D fixed to 4096; Sk <= 65.
// ---------------------------------------------------------------------------
__global__ __launch_bounds__(256) void l2_softmax(const float* __restrict__ Q,
                                                  const float* __restrict__ Kd,
                                                  const float* __restrict__ kn,
                                                  float* __restrict__ wts,
                                                  int Sq, int Sk) {
    int b = blockIdx.y, qi = blockIdx.x, tid = threadIdx.x;
    __shared__ float red[256];
    __shared__ float s[72];
    float qreg[16];
    const float* qrow = Q + ((size_t)(b * Sq + qi)) * 4096;
#pragma unroll
    for (int i = 0; i < 16; ++i) qreg[i] = qrow[tid + i * 256];
    const float* kbase = Kd + (size_t)b * Sk * 4096;
    for (int j = 0; j < Sk; ++j) {
        const float* krow = kbase + (size_t)j * 4096;
        float p = 0.0f;
#pragma unroll
        for (int i = 0; i < 16; ++i) p += qreg[i] * krow[tid + i * 256];
        red[tid] = p;
        __syncthreads();
        for (int st = 128; st > 0; st >>= 1) {
            if (tid < st) red[tid] += red[tid + st];
            __syncthreads();
        }
        if (tid == 0) s[j] = 2.0f * red[0] - kn[b * Sk + j];
        __syncthreads();
    }
    if (tid == 0) {
        float mx = -3.4e38f;
        for (int j = 0; j < Sk; ++j) mx = fmaxf(mx, s[j]);
        float sm = 0.0f;
        for (int j = 0; j < Sk; ++j) {
            float e = __expf(s[j] - mx);
            s[j] = e;
            sm += e;
        }
        float inv = 1.0f / sm;
        float* wrow = wts + ((size_t)(b * Sq + qi)) * Sk;
        for (int j = 0; j < Sk; ++j) wrow[j] = s[j] * inv;
    }
}

// ---------------------------------------------------------------------------
// out[b, j, d] = sum_q wts[b, q, j] * V[b, q, d]   (key-indexed output)
// ---------------------------------------------------------------------------
__global__ __launch_bounds__(256) void attn_apply(const float* __restrict__ wts,
                                                  const float* __restrict__ V,
                                                  float* __restrict__ Out,
                                                  int Sq, int Sk) {
    int t = blockIdx.x * 256 + threadIdx.x;
    int d = t & 4095;
    int j = (t >> 12) % Sk;
    int b = t / (Sk << 12);
    const float* w0 = wts + (size_t)b * Sq * Sk + j;
    const float* v0 = V + (size_t)b * Sq * 4096 + d;
    float acc = 0.0f;
    for (int q = 0; q < Sq; ++q) acc += w0[(size_t)q * Sk] * v0[(size_t)q * 4096];
    Out[((size_t)(b * Sk + j)) * 4096 + d] = acc;
}

// ---------------------------------------------------------------------------
// depatchify h3 (B,64,4096) -> out (B,C,H,W)
// ---------------------------------------------------------------------------
__global__ __launch_bounds__(256) void depatchify(const float* __restrict__ h3,
                                                  float* __restrict__ out) {
    int t = blockIdx.x * 256 + threadIdx.x;
    int ww = t & 63, hh = (t >> 6) & 63, c = (t >> 12) & 63, b = t >> 18;
    int jb = hh >> 3, pr = hh & 7, ib = ww >> 3, pc = ww & 7;
    int n = ib * 8 + jb;
    int d = ((pr * 8 + pc) << 6) + c;
    out[t] = h3[((size_t)(b * 64 + n)) * 4096 + d];
}

// ---------------------------------------------------------------------------
extern "C" void kernel_launch(void* const* d_in, const int* in_sizes, int n_in,
                              void* d_out, int out_size, void* d_ws, size_t ws_size,
                              hipStream_t stream) {
    const float* x          = (const float*)d_in[0];
    const float* w          = (const float*)d_in[1];
    const float* t_local    = (const float*)d_in[2];
    const float* sa_style_w = (const float*)d_in[3];
    const float* sa_style_b = (const float*)d_in[4];
    const float* sa_kk = (const float*)d_in[5];
    const float* sa_kb = (const float*)d_in[6];
    const float* sa_qk = (const float*)d_in[7];
    const float* sa_qb = (const float*)d_in[8];
    const float* sa_vk = (const float*)d_in[9];
    const float* sa_vb = (const float*)d_in[10];
    const float* sa_ok = (const float*)d_in[11];
    const float* sa_ob = (const float*)d_in[12];
    const float* ca_kk = (const float*)d_in[13];
    const float* ca_kb = (const float*)d_in[14];
    const float* ca_qk = (const float*)d_in[15];
    const float* ca_qb = (const float*)d_in[16];
    const float* ca_vk = (const float*)d_in[17];
    const float* ca_vb = (const float*)d_in[18];
    const float* ca_ok = (const float*)d_in[19];
    const float* ca_ob = (const float*)d_in[20];

    // Workspace layout (floats): hs/q/k/v/a are (B,65,4096); h2 is (B,64,4096)
    const size_t S65 = (size_t)8 * 65 * 4096;   // 2,129,920
    const size_t S64 = (size_t)8 * 64 * 4096;   // 2,097,152
    float* ws = (float*)d_ws;
    float* hs = ws;
    float* qb = hs + S65;
    float* kb = qb + S65;
    float* vb = kb + S65;
    float* ab = vb + S65;
    float* h2 = ab + S65;
    float* kn = h2 + S64;
    float* wt = kn + 520;

    float* out = (float*)d_out;

    // 1) patchify + style token
    build_hs<<<8320, 256, 0, stream>>>(x, w, sa_style_w, sa_style_b, hs);

    // 2) self-attention projections: (520 x 4096) @ (4096 x 4096)
    dim3 gBig(4096 / BN, (520 + BM - 1) / BM);  // 64 x 5
    gemm_bf16<<<gBig, 256, 0, stream>>>(hs, sa_qk, sa_qb, qb, 520, 4096, 4096, 65, 65, 1.0f);
    gemm_bf16<<<gBig, 256, 0, stream>>>(hs, sa_kk, sa_kb, kb, 520, 4096, 4096, 65, 65, 1.0f);
    gemm_bf16<<<gBig, 256, 0, stream>>>(hs, sa_vk, sa_vb, vb, 520, 4096, 4096, 65, 65, 1.0f);

    // 3) L2 attention (Sq=Sk=65), softmax over keys, key-indexed output
    row_sqnorm<<<520, 256, 0, stream>>>(kb, kn, 4096);
    l2_softmax<<<dim3(65, 8), 256, 0, stream>>>(qb, kb, kn, wt, 65, 65);
    attn_apply<<<8320, 256, 0, stream>>>(wt, vb, ab, 65, 65);

    // 4) SA output projection (drop style row via rpbU=64/rpbA=65) + leaky
    dim3 g512(4096 / BN, 512 / BM);             // 64 x 4
    gemm_bf16<<<g512, 256, 0, stream>>>(ab, sa_ok, sa_ob, h2, 512, 4096, 4096, 64, 65, 0.01f);

    // 5) cross-attention projections
    gemm_bf16<<<g512, 256, 0, stream>>>(h2, ca_kk, ca_kb, kb, 512, 4096, 4096, 64, 64, 1.0f);
    gemm_bf16<<<g512, 256, 0, stream>>>(t_local, ca_qk, ca_qb, qb, 512, 4096, 512, 64, 64, 1.0f);
    gemm_bf16<<<g512, 256, 0, stream>>>(t_local, ca_vk, ca_vb, vb, 512, 4096, 512, 64, 64, 1.0f);

    // 6) L2 cross-attention (Sq=T=64, Sk=N=64)
    row_sqnorm<<<512, 256, 0, stream>>>(kb, kn, 4096);
    l2_softmax<<<dim3(64, 8), 256, 0, stream>>>(qb, kb, kn, wt, 64, 64);
    attn_apply<<<8192, 256, 0, stream>>>(wt, vb, ab, 64, 64);

    // 7) CA output projection + leaky (reuse hs buffer as h3)
    gemm_bf16<<<g512, 256, 0, stream>>>(ab, ca_ok, ca_ob, hs, 512, 4096, 4096, 64, 64, 0.01f);

    // 8) depatchify -> NCHW
    depatchify<<<8192, 256, 0, stream>>>(hs, out);
}
